// risk_interaction_42863773614500
// MI455X (gfx1250) — matrix operand order repaired and checked by reading, not compiled, not run on hardware
//
#include <hip/hip_runtime.h>

// ---------------------------------------------------------------------------
// MI455X (gfx1250, wave32) implementation.
//
// Dominant cost: 31*256*256 ~= 2.03M pairs, each through a 4->128->64->1 MLP
// (~36 GFLOP) -> compute bound (output is only 8MB vs 23.3 TB/s HBM).
// Strategy: 16 pairs per wave as WMMA M-dim, f16 A/B with f32 accumulate:
//   layer1: 8x  v_wmma_f32_16x16x32_f16 (K=4 zero-padded to 32)
//   layer2: 16x v_wmma_f32_16x16x32_f16 (K=128 as 4 chunks, N=64 as 4 tiles)
//   layer3: VALU dot (N=1) + one cross-half shuffle reduce
// Weights pre-packed once into WMMA B-fragment layout (L2-resident).
// ---------------------------------------------------------------------------

typedef __attribute__((ext_vector_type(16))) _Float16 v16h;
typedef __attribute__((ext_vector_type(8)))  float    v8f;

#define HALF_PI_F 1.57079632679489661923f

__device__ __forceinline__ v8f wmma16x16x32(v16h a, v16h b, v8f c) {
  // (neg_a, A, neg_b, B, c_mod, C, reuse_a, reuse_b)
  return __builtin_amdgcn_wmma_f32_16x16x32_f16(false, a, false, b,
                                                (short)0, c, false, false);
}

// ---------------------------------------------------------------------------
// Kernel A: per (t,n) geometry: cur, v, ang.   a layout: (N=256, 2, T=32)
// ---------------------------------------------------------------------------
__global__ __launch_bounds__(256) void prep_geo_kernel(
    const float* __restrict__ a, float* __restrict__ curx,
    float* __restrict__ cury, float* __restrict__ vel,
    float* __restrict__ angv) {
  const int t = blockIdx.x;    // 0..30
  const int n = threadIdx.x;   // 0..255
  const float cx = a[n * 64 + (t + 1)];
  const float cy = a[n * 64 + 32 + (t + 1)];
  const float px = a[n * 64 + t];
  const float py = a[n * 64 + 32 + t];
  const float dx = cx - px, dy = cy - py;
  const float v = sqrtf(dx * dx + dy * dy + 1e-12f) * 2.0f;  // / 0.5
  const float ang = (dx == 0.0f) ? HALF_PI_F : atanf(dy / dx);
  const int o = t * 256 + n;
  curx[o] = cx; cury[o] = cy; vel[o] = v; angv[o] = ang;
}

// ---------------------------------------------------------------------------
// Kernel W: pack pairwise-MLP weights into WMMA B-fragment layout (f16).
// B (32x16) contiguous k order used here: lanes 0-15 hold K=0-15 (2 K per VGPR),
// lanes 16-31 hold K=16-31; lane's column n = lane & 15.
// Fragment storage: frag[tile][lane][e], so a wave loads v16h at lane*16.
// ---------------------------------------------------------------------------
__global__ __launch_bounds__(256) void prep_frags_kernel(
    const float* __restrict__ W1,   // gW1[1]: (4,128)
    const float* __restrict__ W2,   // gW2[1]: (128,64)
    _Float16* __restrict__ b1f,     // 8 tiles  * 512
    _Float16* __restrict__ b2f) {   // 16 tiles * 512
  const int idx = blockIdx.x * 256 + threadIdx.x;
  const int stride = gridDim.x * 256;
  // layer1: 8 N-tiles over K-window [0,32); rows >= 4 are zero padding.
  for (int q = idx; q < 8 * 512; q += stride) {
    const int tile = q >> 9, lane = (q >> 4) & 31, e = q & 15;
    float val = 0.0f;
    if (lane < 16) {                       // k = e   (lanes>=16: k=16+e >= 4)
      if (e < 4) val = W1[e * 128 + tile * 16 + lane];
    }
    b1f[q] = (_Float16)val;
  }
  // layer2: tile = kc*4 + nt over (128 x 64).
  for (int q = idx; q < 16 * 512; q += stride) {
    const int tile = q >> 9, lane = (q >> 4) & 31, e = q & 15;
    const int kc = tile >> 2, nt = tile & 3;
    const int k = kc * 32 + e + ((lane >= 16) ? 16 : 0);
    const int n = nt * 16 + (lane & 15);
    b2f[q] = (_Float16)W2[k * 64 + n];
  }
}

// ---------------------------------------------------------------------------
// Scalar streaming MLP: IN -> 128 -> 64 -> 1 with PReLU.  W1:(IN,128),
// W2:(128,64), W3:(64).  64 f32 accumulators stay in VGPRs.
// ---------------------------------------------------------------------------
template <int IN>
__device__ float mlp3_scalar(const float* x, const float* __restrict__ W1,
                             const float* __restrict__ b1,
                             const float* __restrict__ W2,
                             const float* __restrict__ b2,
                             const float* __restrict__ W3,
                             const float* __restrict__ b3,
                             const float* __restrict__ alpha) {
  float acc[64];
#pragma unroll
  for (int o = 0; o < 64; ++o) acc[o] = b2[o];
  for (int k = 0; k < 128; ++k) {
    float h = b1[k];
#pragma unroll
    for (int ii = 0; ii < IN; ++ii) h += x[ii] * W1[ii * 128 + k];
    h = (h >= 0.0f) ? h : alpha[0] * h;
    const float* w2r = W2 + k * 64;
#pragma unroll
    for (int o = 0; o < 64; ++o) acc[o] += h * w2r[o];
  }
  float s = b3[0];
#pragma unroll
  for (int o = 0; o < 64; ++o) {
    float h = acc[o];
    h = (h >= 0.0f) ? h : alpha[1] * h;
    s += h * W3[o];
  }
  return (s >= 0.0f) ? s : alpha[2] * s;
}

// ---------------------------------------------------------------------------
// Kernel B: per-node feature MLPs + node MLP (1.5% of FLOPs -> scalar VALU).
// ---------------------------------------------------------------------------
__global__ __launch_bounds__(256) void node_mlp_kernel(
    const float* __restrict__ curx, const float* __restrict__ cury,
    const float* __restrict__ vel, const float* __restrict__ angv,
    float* __restrict__ nodev,
    const float* __restrict__ fW1, const float* __restrict__ fb1,
    const float* __restrict__ fW2, const float* __restrict__ fb2,
    const float* __restrict__ fW3, const float* __restrict__ fb3,
    const float* __restrict__ falpha,
    const float* __restrict__ gW1, const float* __restrict__ gb1,
    const float* __restrict__ gW2, const float* __restrict__ gb2,
    const float* __restrict__ gW3, const float* __restrict__ gb3,
    const float* __restrict__ galpha) {
  const int t = blockIdx.x, n = threadIdx.x;
  const int o = t * 256 + n, ol = t * 256 + 255;
  const float cx = curx[o], cy = cury[o], v = vel[o], ang = angv[o];
  const float lx = curx[ol], ly = cury[ol], vl = vel[ol], al = angv[ol];
  const float ddx = cx - lx, ddy = cy - ly;
  const float dneib = sqrtf(ddx * ddx + ddy * ddy + 1e-12f);
  const float mf = (dneib <= 12.0f) ? 1.0f : 0.0f;

  float fin[4][2] = {{cx, mf * lx}, {cy, mf * ly}, {v, mf * vl}, {ang, mf * al}};
  float pf[4];
#pragma unroll
  for (int f = 0; f < 4; ++f)
    pf[f] = mlp3_scalar<2>(fin[f], fW1 + f * 256, fb1 + f * 128,
                           fW2 + f * 8192, fb2 + f * 64, fW3 + f * 64,
                           fb3 + f, falpha + f * 3);
  nodev[o] = mlp3_scalar<4>(pf, gW1, gb1, gW2, gb2, gW3, gb3, galpha);
}

// ---------------------------------------------------------------------------
// Kernel C: pairwise risk.  8 waves/block; each wave owns 16 pairs sharing
// (t,i), with j = j0..j0+15 (16 | 256 so a wave never crosses a row).
// ---------------------------------------------------------------------------
__global__ __launch_bounds__(256) void risk_pair_kernel(
    const float* __restrict__ curx, const float* __restrict__ cury,
    const float* __restrict__ vel, const float* __restrict__ angv,
    const float* __restrict__ nodev, const float* __restrict__ clu,
    const _Float16* __restrict__ b1f, const _Float16* __restrict__ b2f,
    const float* __restrict__ gb1, const float* __restrict__ gb2,
    const float* __restrict__ gW3, const float* __restrict__ gb3,
    const float* __restrict__ galpha, float* __restrict__ out) {
  __shared__ float s_b1[128];
  __shared__ float s_b2[64];
  __shared__ float s_w3[64];
  __shared__ float s_misc[4];
  // per-wave staging: h1 as f16 [16][132] (4224B) / reused as h2 f32 [16][66]
  __shared__ __align__(16) unsigned char stage[8][4352];

  const int tid = threadIdx.x;
  if (tid < 128) s_b1[tid] = gb1[128 + tid];
  if (tid < 64) s_b2[tid] = gb2[64 + tid];
  if (tid < 64) s_w3[tid] = gW3[64 + tid];
  if (tid < 3) s_misc[tid] = galpha[3 + tid];
  if (tid == 3) s_misc[3] = gb3[1];
  __syncthreads();

  const int lane = tid & 31;
  const int wave = tid >> 5;
  const int base = (blockIdx.x * 8 + wave) << 4;  // first pair index
  const int t = base >> 16;
  const int i = (base >> 8) & 255;
  const int j0 = base & 255;
  const int m = lane & 15;
  const bool hi = lane >= 16;
  const int j = j0 + m;

  const float alpha0 = s_misc[0], alpha1 = s_misc[1], alpha2 = s_misc[2];
  const float bias3 = s_misc[3];

  const float node_i = nodev[t * 256 + i];
  const float node_j = nodev[t * 256 + j];
  const float clu_i = clu[i];
  const float clu_j = clu[j];

  // ---- layer 1: [16x4 (pad 32)] x [4x128] via 8 WMMAs -------------------
  // A layout (16-bit, 16x32): lanes 0-15 rows M=0..15 with K=e (e<8) /
  // K=8+e (e>=8); lanes 16-31 hold K+8 -> all-zero here (K>=4 is padding).
  v16h a1v;
#pragma unroll
  for (int e = 0; e < 16; ++e) a1v[e] = (_Float16)0.0f;
  a1v[0] = (_Float16)(hi ? 0.0f : node_i);
  a1v[1] = (_Float16)(hi ? 0.0f : node_j);
  a1v[2] = (_Float16)(hi ? 0.0f : clu_i);
  a1v[3] = (_Float16)(hi ? 0.0f : clu_j);

  v8f h1acc[8];
#pragma unroll
  for (int nt = 0; nt < 8; ++nt) {
    v16h bfrag = *(const v16h*)(b1f + (nt << 9) + (lane << 4));
    v8f c = {};
    h1acc[nt] = wmma16x16x32(a1v, bfrag, c);
  }

  // bias + PReLU; stage h1 as f16 [m][n], row stride 132 (bank-conflict pad)
  _Float16* h1 = (_Float16*)&stage[wave][0];
  const int mb = hi ? 8 : 0;  // D layout: lanes>=16 hold rows M=8..15
#pragma unroll
  for (int nt = 0; nt < 8; ++nt) {
    const int n = (nt << 4) + m;
    const float bn = s_b1[n];
#pragma unroll
    for (int r = 0; r < 8; ++r) {
      float x = h1acc[nt][r] + bn;
      x = (x >= 0.0f) ? x : alpha0 * x;
      h1[(mb + r) * 132 + n] = (_Float16)x;
    }
  }
  asm volatile("s_wait_dscnt 0" ::: "memory");

  // ---- layer 2: [16x128] x [128x64] via 16 WMMAs ------------------------
  v8f h2acc[4];
#pragma unroll
  for (int nt = 0; nt < 4; ++nt) { v8f z = {}; h2acc[nt] = z; }
  const int hioff = hi ? 16 : 0;
#pragma unroll
  for (int kc = 0; kc < 4; ++kc) {
    // rebuild A fragment from LDS (row m, K window kc*32..+31)
    v16h af;
    const _Float16* row = h1 + m * 132 + (kc << 5) + hioff;
#pragma unroll
    for (int p = 0; p < 8; ++p) {
      const int kb = 2 * p;
      af[2 * p] = row[kb];
      af[2 * p + 1] = row[kb + 1];
    }
#pragma unroll
    for (int nt = 0; nt < 4; ++nt) {
      v16h bfrag = *(const v16h*)(b2f + (((kc << 2) + nt) << 9) + (lane << 4));
      h2acc[nt] = wmma16x16x32(af, bfrag, h2acc[nt]);
    }
  }
  asm volatile("s_wait_dscnt 0" ::: "memory");

  // bias + PReLU; stage h2 as f32 [m][n], row stride 66
  float* h2 = (float*)&stage[wave][0];
#pragma unroll
  for (int nt = 0; nt < 4; ++nt) {
    const int n = (nt << 4) + m;
    const float bn = s_b2[n];
#pragma unroll
    for (int r = 0; r < 8; ++r) {
      float x = h2acc[nt][r] + bn;
      x = (x >= 0.0f) ? x : alpha1 * x;
      h2[(mb + r) * 66 + n] = x;
    }
  }
  asm volatile("s_wait_dscnt 0" ::: "memory");

  // ---- layer 3: dot(h2[m], W3) ; halves split n, then xor-16 reduce -----
  const int nof = hi ? 32 : 0;
  float s = 0.0f;
#pragma unroll
  for (int n = 0; n < 32; ++n) s += h2[m * 66 + nof + n] * s_w3[nof + n];
  s += __shfl_xor(s, 16, 32);
  s += bias3;
  const float bb = (s >= 0.0f) ? s : alpha2 * s;

  // ---- geometry for pair (t, i, j) --------------------------------------
  const float xi = curx[t * 256 + i], yi = cury[t * 256 + i];
  const float xj = curx[t * 256 + j], yj = cury[t * 256 + j];
  const float vi = vel[t * 256 + i], vj = vel[t * 256 + j];
  const float a1 = angv[t * 256 + i], a2 = angv[t * 256 + j];
  const float rx = xj - xi, ry = yj - yi;
  const float dis = sqrtf(rx * rx + ry * ry + 1e-12f);
  const float ang3 = (rx == 0.0f) ? HALF_PI_F : atanf(ry / rx);
  const float lij =
      ((a1 - HALF_PI_F < ang3) && (ang3 < a1 + HALF_PI_F)) ? 1.0f : 0.0f;
  const float vv =
      fabsf(vi * cosf(fabsf(a1 - ang3)) - vj * cosf(fabsf(a2 - ang3)));
  const float risk = (i == j) ? 0.0f : (vv / dis) * bb * lij;

  if (!hi) out[t * 65536 + i * 256 + j] = risk;
}

// ---------------------------------------------------------------------------
extern "C" void kernel_launch(void* const* d_in, const int* in_sizes, int n_in,
                              void* d_out, int out_size, void* d_ws,
                              size_t ws_size, hipStream_t stream) {
  (void)in_sizes; (void)n_in; (void)out_size; (void)ws_size;
  const float* a      = (const float*)d_in[0];
  const float* clu    = (const float*)d_in[1];
  const float* fW1    = (const float*)d_in[2];
  const float* fb1    = (const float*)d_in[3];
  const float* fW2    = (const float*)d_in[4];
  const float* fb2    = (const float*)d_in[5];
  const float* fW3    = (const float*)d_in[6];
  const float* fb3    = (const float*)d_in[7];
  const float* falpha = (const float*)d_in[8];
  const float* gW1    = (const float*)d_in[9];
  const float* gb1    = (const float*)d_in[10];
  const float* gW2    = (const float*)d_in[11];
  const float* gb2    = (const float*)d_in[12];
  const float* gW3    = (const float*)d_in[13];
  const float* gb3    = (const float*)d_in[14];
  const float* galpha = (const float*)d_in[15];
  float* out = (float*)d_out;

  // workspace: 5 x (31*256 f32) geometry arrays + packed f16 B-fragments
  char* ws = (char*)d_ws;
  float* curx  = (float*)(ws + 0 * 31744);
  float* cury  = (float*)(ws + 1 * 31744);
  float* vel   = (float*)(ws + 2 * 31744);
  float* angv  = (float*)(ws + 3 * 31744);
  float* nodev = (float*)(ws + 4 * 31744);
  _Float16* b1f = (_Float16*)(ws + 5 * 31744);          // 4096 halfs (8 KB)
  _Float16* b2f = (_Float16*)(ws + 5 * 31744 + 8192);   // 8192 halfs (16 KB)

  prep_geo_kernel<<<31, 256, 0, stream>>>(a, curx, cury, vel, angv);
  prep_frags_kernel<<<16, 256, 0, stream>>>(gW1 + 512, gW2 + 8192, b1f, b2f);
  node_mlp_kernel<<<31, 256, 0, stream>>>(curx, cury, vel, angv, nodev,
                                          fW1, fb1, fW2, fb2, fW3, fb3, falpha,
                                          gW1, gb1, gW2, gb2, gW3, gb3, galpha);
  // 31*256*256 pairs / (8 waves * 16 pairs) = 15872 blocks
  risk_pair_kernel<<<15872, 256, 0, stream>>>(curx, cury, vel, angv, nodev, clu,
                                              b1f, b2f, gb1, gb2, gW3, gb3,
                                              galpha, out);
}
